// SparseCodingFista_41807211659992
// MI455X (gfx1250) — matrix operand order repaired and checked by reading, not compiled, not run on hardware
//
#include <hip/hip_runtime.h>
#include <hip/hip_bf16.h>

typedef __attribute__((ext_vector_type(16))) __bf16 v16bf;
typedef __attribute__((ext_vector_type(8)))  __bf16 v8bf;
typedef __attribute__((ext_vector_type(8)))  float  v8f;

#define LAMBDAV 0.1f
#define NSTEPS  100
#define PITERS  50
#define DIN     64
#define DOUT    256
#define BTOT    16384

// ---------------------------------------------------------------------------
// Kernel 1: power iteration for L = lambda_max(2 W W^T); emit invL, thr, and
// G = 2 W W^T in bf16 (row-major, dense) to workspace. Q is never
// materialized: Qv = 2 * W (W^T v).
// ---------------------------------------------------------------------------
__global__ __launch_bounds__(256) void fista_prep(const float* __restrict__ W,
                                                  float* __restrict__ scal,
                                                  __bf16* __restrict__ Gws) {
  __shared__ float Wl[DOUT * DIN];   // 64 KB
  __shared__ float v[DOUT];
  __shared__ float vn[DOUT];
  __shared__ float w64[DIN];
  __shared__ float red[DOUT];
  const int tid = threadIdx.x;

  for (int i = tid; i < DOUT * DIN; i += 256) Wl[i] = W[i];
  v[tid] = 1.0f / 16.0f;             // 1/sqrt(256)
  __syncthreads();

  for (int it = 0; it < PITERS; ++it) {
    if (tid < DIN) {
      float s = 0.f;
      for (int i = 0; i < DOUT; ++i) s += v[i] * Wl[i * DIN + tid];
      w64[tid] = s;
    }
    __syncthreads();
    float s = 0.f;
    for (int k = 0; k < DIN; ++k) s += Wl[tid * DIN + k] * w64[k];
    float q = 2.f * s;
    vn[tid] = q;
    red[tid] = q * q;
    __syncthreads();
    for (int st = 128; st > 0; st >>= 1) {
      if (tid < st) red[tid] += red[tid + st];
      __syncthreads();
    }
    v[tid] = vn[tid] * (1.0f / sqrtf(red[0]));
    __syncthreads();
  }

  // L = v^T Q v = 2 * ||W^T v||^2
  if (tid < DIN) {
    float s = 0.f;
    for (int i = 0; i < DOUT; ++i) s += v[i] * Wl[i * DIN + tid];
    w64[tid] = s;
  }
  __syncthreads();
  red[tid] = (tid < DIN) ? w64[tid] * w64[tid] : 0.f;
  __syncthreads();
  for (int st = 128; st > 0; st >>= 1) {
    if (tid < st) red[tid] += red[tid + st];
    __syncthreads();
  }
  if (tid == 0) {
    float L = 2.f * red[0];
    scal[0] = 1.f / L;           // invL
    scal[1] = LAMBDAV / L;       // thr
  }

  // G = 2 W W^T, bf16, dense row-major
  for (int idx = tid; idx < DOUT * DOUT; idx += 256) {
    int i = idx >> 8, j = idx & 255;
    float s = 0.f;
    for (int k = 0; k < DIN; ++k) s += Wl[i * DIN + k] * Wl[j * DIN + k];
    Gws[idx] = (__bf16)(2.f * s);
  }
}

// ---------------------------------------------------------------------------
// Kernel 2: persistent FISTA. Block = 128 threads (4 waves), 64 rows/block.
// Each wave owns a 16-row M-tile; all 100 iterations run on-chip.
//   grad = y @ G  (bf16 WMMA, f32 accum);  u = y - invL*grad + c*invL
//   x2 = prox(u, thr);  y = x2 + beta*(x2 - X);  X = x2
// State: y, acc in VGPRs; X, c in LDS (f32); y staged bf16 in LDS for WMMA A.
// ---------------------------------------------------------------------------
#define GSTR 264   // bf16 row stride (528 B: 16B aligned, bank-spread)
#define CSTR 260   // f32 row stride (conflict-free half-wave split)

static __device__ __forceinline__ v16bf mk16(v8bf lo, v8bf hi) {
  v16bf r;
#pragma unroll
  for (int i = 0; i < 8; ++i) { r[i] = lo[i]; r[i + 8] = hi[i]; }
  return r;
}

__global__ __launch_bounds__(128) void fista_main(const float* __restrict__ inputs,
                                                  const float* __restrict__ Wg,
                                                  const float* __restrict__ X0,
                                                  const float* __restrict__ scal,
                                                  const __bf16* __restrict__ Gws,
                                                  float* __restrict__ Xout) {
  __shared__ __bf16 Gs[DOUT * GSTR];     // 135168 B  (init reuse: W as f32)
  __shared__ float  cs[64 * CSTR];       //  66560 B
  __shared__ float  Xs[64 * CSTR];       //  66560 B
  __shared__ __bf16 ys[4 * 16 * GSTR];   //  33792 B  (init reuse: inputs as f32)

  const int tid  = threadIdx.x;
  const int lane = tid & 31;
  const int w    = tid >> 5;
  const int b0   = blockIdx.x * 64;
  const float invL = scal[0];
  const float thr  = scal[1];

  // ---- init phase: alias LDS as f32 scratch ----
  float* Wf  = reinterpret_cast<float*>(Gs);   // 256*64 f32
  float* inf = reinterpret_cast<float*>(ys);   // 64*64  f32
  for (int i = tid; i < DOUT * DIN; i += 128) Wf[i] = Wg[i];
  for (int i = tid; i < 64 * DIN; i += 128) inf[i] = inputs[(size_t)b0 * DIN + i];
  __syncthreads();

  const float c2 = 2.f * invL;
  for (int idx = tid; idx < 64 * DOUT; idx += 128) {
    int r = idx >> 8, cc = idx & 255;
    float s = 0.f;
#pragma unroll
    for (int k = 0; k < DIN; ++k) s += inf[r * DIN + k] * Wf[cc * DIN + k];
    cs[r * CSTR + cc] = c2 * s;          // c * invL
  }
  __syncthreads();

  // load G (padded stride); overwrite W scratch
  for (int idx = tid; idx < DOUT * DOUT; idx += 128)
    Gs[(idx >> 8) * GSTR + (idx & 255)] = Gws[idx];

  // ---- load X0: X -> LDS, y -> regs, y0 staged bf16 ----
  const int mrow = (lane >> 4) * 8;      // C/D row half
  const int ncl  = lane & 15;            // C/D column within tile
  const int rowLoc = w * 16;
  v8f yv[16];
#pragma unroll
  for (int nt = 0; nt < 16; ++nt) {
#pragma unroll
    for (int r = 0; r < 8; ++r) {
      int rl = rowLoc + mrow + r;
      float x = X0[(size_t)(b0 + rl) * DOUT + nt * 16 + ncl];
      yv[nt][r] = x;
      Xs[rl * CSTR + nt * 16 + ncl] = x;
      ys[(size_t)w * 16 * GSTR + (mrow + r) * GSTR + nt * 16 + ncl] = (__bf16)x;
    }
  }
  __syncthreads();

  // fragment addressing
  const __bf16* ysw = ys + (size_t)w * 16 * GSTR;
  const int am   = lane & 15;                 // A row
  const int aoff = (lane >> 4) ? 16 : 0;      // A: K 0..15 / 16..31
  const int boff = (lane >> 4) ? 16 : 0;      // B: K 0..15 / 16..31 (G symmetric -> row fetch)
  const __bf16* apBase = ysw + am * GSTR + aoff;
  const __bf16* bpBase = Gs + ncl * GSTR + boff;

  float t = 1.f;
#pragma unroll 1
  for (int it = 0; it < NSTEPS; ++it) {
    const v8f z = {0.f, 0.f, 0.f, 0.f, 0.f, 0.f, 0.f, 0.f};
    v8f acc[16];
#pragma unroll
    for (int nt = 0; nt < 16; ++nt) acc[nt] = z;

    // ---- grad = y @ G : 2-deep software-pipelined operand stream ----
    // linear fragment index f = kc*16 + nt; B fragment address:
    //   bpBase + (f&15)*16*GSTR + (f>>4)*32
    v8bf alo = *(const v8bf*)(apBase);
    v8bf ahi = *(const v8bf*)(apBase + 8);
    v8bf anlo, anhi;                       // next-kc A fragment
    v8bf bl[2], bh[2];
    bl[0] = *(const v8bf*)(bpBase);
    bh[0] = *(const v8bf*)(bpBase + 8);
    {
      const __bf16* p1 = bpBase + 16 * GSTR;   // f = 1
      bl[1] = *(const v8bf*)(p1);
      bh[1] = *(const v8bf*)(p1 + 8);
    }
    v16bf a = mk16(alo, ahi);
#pragma unroll
    for (int f = 0; f < 128; ++f) {
      const int kc = f >> 4;
      const int nt = f & 15;
      // prefetch next-kc A fragment at each kc-block start (16 WMMAs of cover)
      if (nt == 0 && kc < 7) {
        anlo = *(const v8bf*)(apBase + (kc + 1) * 32);
        anhi = *(const v8bf*)(apBase + (kc + 1) * 32 + 8);
      }
      v16bf b = mk16(bl[f & 1], bh[f & 1]);
      // prefetch B fragment f+2 into the buffer just freed (2 WMMAs of cover)
      if (f + 2 < 128) {
        const int f2 = f + 2;
        const __bf16* nbp = bpBase + (f2 & 15) * 16 * GSTR + (f2 >> 4) * 32;
        bl[f & 1] = *(const v8bf*)(nbp);
        bh[f & 1] = *(const v8bf*)(nbp + 8);
      }
      acc[nt] = __builtin_amdgcn_wmma_f32_16x16x32_bf16(
          false, a, false, b, (short)0, acc[nt], false, false);
      // commit next A fragment at kc-block end
      if (nt == 15 && kc < 7) a = mk16(anlo, anhi);
    }

    float t2   = 0.5f + 0.5f * sqrtf(1.f + 4.f * t * t);
    float beta = (t - 1.f) / t2;
    t = t2;

    // ---- prox + momentum update (f32), restage y as bf16 ----
#pragma unroll
    for (int nt = 0; nt < 16; ++nt) {
#pragma unroll
      for (int r = 0; r < 8; ++r) {
        int rl = rowLoc + mrow + r;
        int ci = rl * CSTR + nt * 16 + ncl;
        float cval = cs[ci];
        float xold = Xs[ci];
        float u  = yv[nt][r] - invL * acc[nt][r] + cval;
        float x2 = fmaxf(u - thr, 0.f) + fminf(u + thr, 0.f);
        float yn = x2 + beta * (x2 - xold);
        Xs[ci] = x2;
        yv[nt][r] = yn;
        ys[(size_t)w * 16 * GSTR + (mrow + r) * GSTR + nt * 16 + ncl] = (__bf16)yn;
      }
    }
  }

  // ---- write X ----
#pragma unroll
  for (int nt = 0; nt < 16; ++nt) {
#pragma unroll
    for (int r = 0; r < 8; ++r) {
      int rl = rowLoc + mrow + r;
      Xout[(size_t)(b0 + rl) * DOUT + nt * 16 + ncl] = Xs[rl * CSTR + nt * 16 + ncl];
    }
  }
}

// ---------------------------------------------------------------------------
extern "C" void kernel_launch(void* const* d_in, const int* in_sizes, int n_in,
                              void* d_out, int out_size, void* d_ws, size_t ws_size,
                              hipStream_t stream) {
  const float* inputs = (const float*)d_in[0];   // [16384, 64]
  const float* W      = (const float*)d_in[1];   // [256, 64]
  const float* X0     = (const float*)d_in[2];   // [16384, 256]

  float*  scal = (float*)d_ws;                         // [invL, thr]
  __bf16* Gws  = (__bf16*)((char*)d_ws + 256);         // 256*256 bf16

  fista_prep<<<1, 256, 0, stream>>>(W, scal, Gws);
  fista_main<<<BTOT / 64, 128, 0, stream>>>(inputs, W, X0, scal, Gws, (float*)d_out);
}
